// BandSelector_36455682408889
// MI455X (gfx1250) — compile-verified, matrix operand order checked by reading, not run on hardware
//
#include <hip/hip_runtime.h>
#include <hip/hip_bf16.h>

// ---------------------------------------------------------------------------
// BandSelector for MI455X (gfx1250).
//   M = B*C = 1792, K = H*W = 16384, N = HID = 256
//   scores = relu(feat @ w1 + b1) @ w2 + b2   -> top3 -> gather bands
// GEMM runs on v_wmma_f32_16x16x32_bf16 (f32 accumulate) to stay memory-bound.
// ---------------------------------------------------------------------------

typedef __bf16 v16bf __attribute__((ext_vector_type(16)));
typedef __bf16 v8bf  __attribute__((ext_vector_type(8)));
typedef __bf16 v4bf  __attribute__((ext_vector_type(4)));
typedef float  v8f   __attribute__((ext_vector_type(8)));

#define BATCH 256
#define BANDS 7
#define KTOT  16384      // H*W
#define HID   256
#define MROWS (BATCH * BANDS)   // 1792 = 112 * 16
#define SEL   3
#define CHUNK 128               // K elements staged per LDS round
#define ASTRIDE 136             // LDS row stride in bf16 (272B, 16B aligned, bank-staggered)

// ---------------------------------------------------------------------------
// Phase 1: w1 (16384 x 256, f32, row-major [n][h]) -> bf16 transposed [h][n].
// LDS-tiled 32x32 transpose so both global sides stay coalesced.
// ---------------------------------------------------------------------------
__global__ __launch_bounds__(256) void cvt_transpose_w1(
    const float* __restrict__ w1, __bf16* __restrict__ out) {
  __shared__ float tile[32][33];
  const int n0 = blockIdx.x * 32;   // K (pixel) tile, 512 tiles
  const int h0 = blockIdx.y * 32;   // HID tile, 8 tiles
  const int tx = threadIdx.x;       // 0..31
  const int ty = threadIdx.y;       // 0..7
#pragma unroll
  for (int i = 0; i < 32; i += 8)
    tile[ty + i][tx] = w1[(size_t)(n0 + ty + i) * HID + (h0 + tx)];
  __syncthreads();
#pragma unroll
  for (int i = 0; i < 32; i += 8)
    out[(size_t)(h0 + ty + i) * KTOT + (n0 + tx)] = (__bf16)tile[tx][ty + i];
}

// ---------------------------------------------------------------------------
// Phase 2: fused GEMM + score.
// Grid: 112 WGs (one 16-row M-tile each -> x streamed from HBM exactly once).
// Block: 512 threads = 16 waves; wave w owns output N-columns [16w, 16w+16).
// A chunk (16 x 128 f32) is converted to bf16 in LDS cooperatively; each wave
// then runs 4 wmma steps per chunk (2x ds_load_b128 A-frag, 2x global b128
// B-frag per step). Epilogue: relu(acc+b1)*w2 reduced over N via ds_add_f32.
// ---------------------------------------------------------------------------
__global__ __launch_bounds__(512) void gemm_score(
    const float*  __restrict__ x,      // (M, K) f32
    const __bf16* __restrict__ wB,     // (N, K) bf16  (transposed w1)
    const float*  __restrict__ b1,     // (256,)
    const float*  __restrict__ w2,     // (256,)
    const float*  __restrict__ b2,     // (1,)
    float*        __restrict__ scores) // (M,)
{
  __shared__ __bf16 As[16 * ASTRIDE];
  __shared__ float  sc[16];

  const int tid   = threadIdx.x;
  const int wave  = tid >> 5;
  const int lane  = tid & 31;
  const int mrow  = lane & 15;   // A row / C column within tile
  const int half  = lane >> 4;   // K-quarter selector (ISA lane split)
  const int Mbase = blockIdx.x * 16;
  const int ncol  = wave * 16 + mrow;        // this lane's output column (C layout)

  if (tid < 16) sc[tid] = 0.0f;

  const __bf16* bptr = wB + (size_t)ncol * KTOT;
  v8f acc = {};

  // cooperative staging coords: each wave stages one row of the 16x128 chunk
  const int srow = tid >> 5;          // 0..15
  const int scol = (tid & 31) * 4;    // 0..124 step 4

  for (int k0 = 0; k0 < KTOT; k0 += CHUNK) {
    // ---- stage A chunk (f32 -> bf16) ----
    {
      const float4 v = *(const float4*)(x + (size_t)(Mbase + srow) * KTOT + k0 + scol);
      v4bf s;
      s[0] = (__bf16)v.x; s[1] = (__bf16)v.y; s[2] = (__bf16)v.z; s[3] = (__bf16)v.w;
      *(v4bf*)&As[srow * ASTRIDE + scol] = s;
    }
    __syncthreads();

#pragma unroll
    for (int j = 0; j < 4; ++j) {
      const int kk = j * 32;
      // A frag: lanes<16 hold K quarters {0..7},{16..23}; lanes>=16 {8..15},{24..31}
      const v8bf alo = *(const v8bf*)&As[mrow * ASTRIDE + kk + half * 8];
      const v8bf ahi = *(const v8bf*)&As[mrow * ASTRIDE + kk + half * 8 + 16];
      const v16bf af = __builtin_shufflevector(alo, ahi,
          0, 1, 2, 3, 4, 5, 6, 7, 8, 9, 10, 11, 12, 13, 14, 15);
      // B frag: same quarter rule, column = ncol, contiguous K in transposed w1
      const v8bf blo = *(const v8bf*)(bptr + k0 + kk + half * 8);
      const v8bf bhi = *(const v8bf*)(bptr + k0 + kk + half * 8 + 16);
      const v16bf bfv = __builtin_shufflevector(blo, bhi,
          0, 1, 2, 3, 4, 5, 6, 7, 8, 9, 10, 11, 12, 13, 14, 15);

      acc = __builtin_amdgcn_wmma_f32_16x16x32_bf16(
          false, af, false, bfv, (short)0, acc, false, false);
    }
    __syncthreads();
  }

  // ---- epilogue: relu(acc + b1[n]) * w2[n], reduce over N into sc[m] ----
  const float b1n = b1[ncol];
  const float w2n = w2[ncol];
#pragma unroll
  for (int r = 0; r < 8; ++r) {
    const int m = r + half * 8;           // C/D layout: lanes>=16 hold M=8..15
    float h = acc[r] + b1n;
    h = h > 0.0f ? h : 0.0f;
    atomicAdd(&sc[m], h * w2n);           // ds_add_f32
  }
  __syncthreads();
  if (tid < 16) scores[Mbase + tid] = sc[tid] + b2[0];
}

// ---------------------------------------------------------------------------
// Phase 3: per-batch top-3 of 7 scores. Strict '>' keeps the lowest index on
// ties (matches lax.top_k). Indices written directly to the output tail.
// ---------------------------------------------------------------------------
__global__ __launch_bounds__(64) void topk3(
    const float* __restrict__ scores, int* __restrict__ idx_out) {
  const int b = blockIdx.x * blockDim.x + threadIdx.x;
  if (b >= BATCH) return;
  float s[BANDS];
#pragma unroll
  for (int c = 0; c < BANDS; ++c) s[c] = scores[b * BANDS + c];
  bool used[BANDS] = {false, false, false, false, false, false, false};
#pragma unroll
  for (int k = 0; k < SEL; ++k) {
    int best = -1;
    float bv = -3.4e38f;
    for (int c = 0; c < BANDS; ++c)
      if (!used[c] && s[c] > bv) { bv = s[c]; best = c; }
    used[best] = true;
    idx_out[b * SEL + k] = best;
  }
}

// ---------------------------------------------------------------------------
// Phase 4: gather selected bands (50 MB copy) in float4 units.
// ---------------------------------------------------------------------------
__global__ __launch_bounds__(256) void gather_bands(
    const float* __restrict__ x, const int* __restrict__ idx,
    float* __restrict__ out) {
  const size_t i = (size_t)blockIdx.x * blockDim.x + threadIdx.x;   // float4 index
  const size_t tot = (size_t)BATCH * SEL * (KTOT / 4);
  if (i >= tot) return;
  const size_t page = i >> 12;            // which (b, s) plane (4096 float4s each)
  const int    off  = (int)(i & 4095);
  const int    b    = (int)(page / SEL);
  const int    s    = (int)(page % SEL);
  const int    band = idx[b * SEL + s];
  const float4 v = ((const float4*)(x + ((size_t)b * BANDS + band) * KTOT))[off];
  ((float4*)out)[i] = v;
}

// ---------------------------------------------------------------------------
extern "C" void kernel_launch(void* const* d_in, const int* in_sizes, int n_in,
                              void* d_out, int out_size, void* d_ws, size_t ws_size,
                              hipStream_t stream) {
  (void)in_sizes; (void)n_in; (void)out_size; (void)ws_size;
  const float* x  = (const float*)d_in[0];
  // d_in[1], d_in[2] = w_qkv / b_qkv: dead in the reference forward.
  const float* w1 = (const float*)d_in[3];
  const float* b1 = (const float*)d_in[4];
  const float* w2 = (const float*)d_in[5];
  const float* b2 = (const float*)d_in[6];

  // workspace: [0, 8MB) bf16 transposed w1 ; then 1792 f32 scores
  __bf16* wsW    = (__bf16*)d_ws;
  float*  scores = (float*)((char*)d_ws + (size_t)HID * KTOT * sizeof(__bf16));

  float* outF = (float*)d_out;
  int*   idxO = (int*)(outF + (size_t)BATCH * SEL * KTOT);  // top_indices tail

  cvt_transpose_w1<<<dim3(KTOT / 32, HID / 32), dim3(32, 8), 0, stream>>>(w1, wsW);
  gemm_score<<<MROWS / 16, 512, 0, stream>>>(x, wsW, b1, w2, b2, scores);
  topk3<<<(BATCH + 63) / 64, 64, 0, stream>>>(scores, idxO);
  gather_bands<<<(unsigned)(((size_t)BATCH * SEL * (KTOT / 4) + 255) / 256), 256, 0,
                 stream>>>(x, idxO, outF);
}
